// Speller_73787538145683
// MI455X (gfx1250) — matrix operand order repaired and checked by reading, not compiled, not run on hardware
//
#include <hip/hip_runtime.h>
#include <hip/hip_bf16.h>

#define E_ 256
#define V_ 1000
#define B_ 64
#define S_ 1024
#define T_ 600
#define NSTEPS (T_ - 1)   // 599
#define VPAD 1024

typedef __attribute__((ext_vector_type(16))) _Float16 v16h;
typedef __attribute__((ext_vector_type(8)))  float    v8f;

__device__ __forceinline__ float sigf_(float x) { return 1.0f / (1.0f + __expf(-x)); }

// ---------------- WMMA fragment loaders (CDNA5 16x16x32 f16 layouts) -------------
// A is MxK row-major f16. Lane L<16: M=L, K runs {k..k+7, k+16..k+23};
// lane L>=16: M=L-16, K runs {k+8..k+15, k+24..k+31}.
__device__ __forceinline__ v16h load_a_frag(const _Float16* __restrict__ A, int lda,
                                            int m0, int kb) {
  int lane = threadIdx.x & 31;
  int m  = m0 + (lane & 15);
  int k0 = kb + ((lane >> 4) << 3);
  const _Float16* p = A + (size_t)m * lda + k0;
  union { float4 f[2]; v16h v; } u;
  u.f[0] = *reinterpret_cast<const float4*>(p);        // K = k0..k0+7
  u.f[1] = *reinterpret_cast<const float4*>(p + 16);   // K = k0+16..k0+23
  return u.v;
}

// On-the-fly f32 -> f16 A fragment; blk points at this lane's row at K-offset kb
// (32-wide K-block fully inside one source segment).
__device__ __forceinline__ v16h cvt_a_frag(const float* __restrict__ blk) {
  int o = ((threadIdx.x & 31) >> 4) << 3;
  const float* p = blk + o;
  float4 x0 = *reinterpret_cast<const float4*>(p);
  float4 x1 = *reinterpret_cast<const float4*>(p + 4);
  float4 y0 = *reinterpret_cast<const float4*>(p + 16);
  float4 y1 = *reinterpret_cast<const float4*>(p + 20);
  v16h v;
  v[0]  = (_Float16)x0.x; v[1]  = (_Float16)x0.y; v[2]  = (_Float16)x0.z; v[3]  = (_Float16)x0.w;
  v[4]  = (_Float16)x1.x; v[5]  = (_Float16)x1.y; v[6]  = (_Float16)x1.z; v[7]  = (_Float16)x1.w;
  v[8]  = (_Float16)y0.x; v[9]  = (_Float16)y0.y; v[10] = (_Float16)y0.z; v[11] = (_Float16)y0.w;
  v[12] = (_Float16)y1.x; v[13] = (_Float16)y1.y; v[14] = (_Float16)y1.z; v[15] = (_Float16)y1.w;
  return v;
}

// B is KxN; C = X @ W^T with W row-major (N,K): column n of B == row n of W.
// Lane L<16: N=L, K runs {kb..kb+7, kb+16..kb+23}; lane L>=16: N=L-16, K runs {kb+8..kb+15, kb+24..kb+31}.
__device__ __forceinline__ v16h load_b_fragT(const _Float16* __restrict__ W, int ldw,
                                             int n0, int kb) {
  int lane = threadIdx.x & 31;
  int n  = n0 + (lane & 15);
  int k0 = kb + ((lane >> 4) << 3);
  const _Float16* p = W + (size_t)n * ldw + k0;
  union { float4 f[2]; v16h v; } u;
  u.f[0] = *reinterpret_cast<const float4*>(p);
  u.f[1] = *reinterpret_cast<const float4*>(p + 16);
  return u.v;
}

#define WMMA_F16(a, b, c) \
  __builtin_amdgcn_wmma_f32_16x16x32_f16(false, (a), false, (b), (short)0, (c), false, false)

// ------------------------- generic WMMA GEMM:  C = A @ W^T + bias ----------------
// 128 threads (4 waves). Wave w: rows m0 = blockIdx.y*64 + w*16; cols n0 = blockIdx.x*64..+63.
__global__ void gemm_wmma_xt(const _Float16* __restrict__ A, int lda,
                             const _Float16* __restrict__ W, int ldw,
                             const float* __restrict__ bias,
                             float* __restrict__ Cf, _Float16* __restrict__ Ch,
                             long ldc, int Nvalid, int K, int relu) {
  int wave = threadIdx.x >> 5;
  int lane = threadIdx.x & 31;
  int m0 = blockIdx.y * 64 + wave * 16;
  int n0 = blockIdx.x * 64;

  v8f acc[4] = {v8f{}, v8f{}, v8f{}, v8f{}};
  for (int k = 0; k < K; k += 32) {
    v16h a = load_a_frag(A, lda, m0, k);
#pragma unroll
    for (int nt = 0; nt < 4; ++nt) {
      v16h b = load_b_fragT(W, ldw, n0 + nt * 16, k);
      acc[nt] = WMMA_F16(a, b, acc[nt]);
    }
  }

  int mbase = m0 + ((lane >> 4) << 3);
#pragma unroll
  for (int nt = 0; nt < 4; ++nt) {
    int n = n0 + nt * 16 + (lane & 15);
    if (n >= Nvalid) continue;
    float bv = bias ? bias[n] : 0.0f;
#pragma unroll
    for (int r = 0; r < 8; ++r) {
      float v = acc[nt][r] + bv;
      if (relu) v = fmaxf(v, 0.0f);
      size_t off = (size_t)(mbase + r) * (size_t)ldc + (size_t)n;
      if (Cf) Cf[off] = v;
      if (Ch) Ch[off] = (_Float16)v;
    }
  }
}

// ---------------- fused LSTM1: gates (WMMA) + cell, A = [emb(y)|ctx|h1old] -------
// grid = 32 blocks (j0 = blockIdx.x*16 over 512), 128 threads (4 waves = 64 batch rows).
// Accumulator g holds gate tile n = g*512 + j0 of the (64 x 2048) gate matrix.
__global__ void lstm1_fused_kernel(const float* __restrict__ embW,
                                   const int* __restrict__ batch_y, int step,
                                   const float* __restrict__ ctx,
                                   const float* __restrict__ h1old,
                                   const _Float16* __restrict__ W1,   // (2048 x 1024)
                                   const float* __restrict__ b1,
                                   float* __restrict__ c1,
                                   float* __restrict__ h1new) {
  int wave = threadIdx.x >> 5;
  int lane = threadIdx.x & 31;
  int m = wave * 16 + (lane & 15);
  int j0 = blockIdx.x * 16;

  int y = (step == 0) ? 0 : batch_y[m * T_ + step];   // SOS = 0
  const float* embrow = embW + (size_t)y * E_;
  const float* ctxrow = ctx + m * E_;
  const float* h1row  = h1old + m * 512;

  v8f acc[4] = {v8f{}, v8f{}, v8f{}, v8f{}};
  for (int kb = 0; kb < 1024; kb += 32) {
    const float* blk = (kb < 256) ? (embrow + kb)
                     : (kb < 512) ? (ctxrow + (kb - 256))
                                  : (h1row + (kb - 512));
    v16h a = cvt_a_frag(blk);
#pragma unroll
    for (int g = 0; g < 4; ++g) {
      v16h b = load_b_fragT(W1, 1024, g * 512 + j0, kb);
      acc[g] = WMMA_F16(a, b, acc[g]);
    }
  }

  int j = j0 + (lane & 15);
  int mbase = wave * 16 + ((lane >> 4) << 3);
  float bi = b1[j], bf = b1[512 + j], bg = b1[1024 + j], bo = b1[1536 + j];
#pragma unroll
  for (int r = 0; r < 8; ++r) {
    int idx = (mbase + r) * 512 + j;
    float i  = sigf_(acc[0][r] + bi);
    float f  = sigf_(acc[1][r] + bf);
    float gg = tanhf(acc[2][r] + bg);
    float o  = sigf_(acc[3][r] + bo);
    float c  = f * c1[idx] + i * gg;
    float h  = o * tanhf(c);
    c1[idx] = c;
    h1new[idx] = h;
  }
}

// ---------------- fused LSTM2: A = [h1new(512)|h2old(256)], K = 768 ---------------
// grid = 16 blocks (j0 over 256), gate tile n = g*256 + j0 of (64 x 1024).
__global__ void lstm2_fused_kernel(const float* __restrict__ h1new,
                                   const float* __restrict__ h2old,
                                   const _Float16* __restrict__ W2,   // (1024 x 768)
                                   const float* __restrict__ b2,
                                   float* __restrict__ c2,
                                   float* __restrict__ h2new) {
  int wave = threadIdx.x >> 5;
  int lane = threadIdx.x & 31;
  int m = wave * 16 + (lane & 15);
  int j0 = blockIdx.x * 16;

  const float* h1row = h1new + m * 512;
  const float* h2row = h2old + m * 256;

  v8f acc[4] = {v8f{}, v8f{}, v8f{}, v8f{}};
  for (int kb = 0; kb < 768; kb += 32) {
    const float* blk = (kb < 512) ? (h1row + kb) : (h2row + (kb - 512));
    v16h a = cvt_a_frag(blk);
#pragma unroll
    for (int g = 0; g < 4; ++g) {
      v16h b = load_b_fragT(W2, 768, g * 256 + j0, kb);
      acc[g] = WMMA_F16(a, b, acc[g]);
    }
  }

  int j = j0 + (lane & 15);
  int mbase = wave * 16 + ((lane >> 4) << 3);
  float bi = b2[j], bf = b2[256 + j], bg = b2[512 + j], bo = b2[768 + j];
#pragma unroll
  for (int r = 0; r < 8; ++r) {
    int idx = (mbase + r) * 256 + j;
    float i  = sigf_(acc[0][r] + bi);
    float f  = sigf_(acc[1][r] + bf);
    float gg = tanhf(acc[2][r] + bg);
    float o  = sigf_(acc[3][r] + bo);
    float c  = f * c2[idx] + i * gg;
    float h  = o * tanhf(c);
    c2[idx] = c;
    h2new[idx] = h;
  }
}

// ---------------- fused cdn stage 1: t = relu([h2|ctx] @ Wt^T + bt) ---------------
// grid = 4 blocks (n0 = blockIdx.x*64 over 256), K = 512; writes f16 t (64 x 256).
__global__ void cdn1_fused_kernel(const float* __restrict__ h2,
                                  const float* __restrict__ ctx,
                                  const _Float16* __restrict__ WtH,  // (256 x 512)
                                  const float* __restrict__ bt,
                                  _Float16* __restrict__ t_h) {
  int wave = threadIdx.x >> 5;
  int lane = threadIdx.x & 31;
  int m = wave * 16 + (lane & 15);
  int n0 = blockIdx.x * 64;

  const float* h2row  = h2 + m * 256;
  const float* ctxrow = ctx + m * 256;

  v8f acc[4] = {v8f{}, v8f{}, v8f{}, v8f{}};
  for (int kb = 0; kb < 512; kb += 32) {
    const float* blk = (kb < 256) ? (h2row + kb) : (ctxrow + (kb - 256));
    v16h a = cvt_a_frag(blk);
#pragma unroll
    for (int nt = 0; nt < 4; ++nt) {
      v16h b = load_b_fragT(WtH, 512, n0 + nt * 16, kb);
      acc[nt] = WMMA_F16(a, b, acc[nt]);
    }
  }

  int mbase = wave * 16 + ((lane >> 4) << 3);
#pragma unroll
  for (int nt = 0; nt < 4; ++nt) {
    int n = n0 + nt * 16 + (lane & 15);
    float bv = bt[n];
#pragma unroll
    for (int r = 0; r < 8; ++r) {
      float v = fmaxf(acc[nt][r] + bv, 0.0f);
      t_h[(mbase + r) * 256 + n] = (_Float16)v;
    }
  }
}

// -------------------------------- setup kernels ----------------------------------
__global__ void mask_cvt_kernel(const float* __restrict__ seq, const int* __restrict__ lens,
                                _Float16* __restrict__ out) {
  size_t idx = (size_t)blockIdx.x * blockDim.x + threadIdx.x;
  if (idx >= (size_t)B_ * S_ * E_) return;
  int s = (int)((idx >> 8) & (S_ - 1));
  int b = (int)(idx >> 18);
  out[idx] = (s < lens[b]) ? (_Float16)seq[idx] : (_Float16)0.0f;
}

__global__ void cvt_kernel(_Float16* __restrict__ dst, const float* __restrict__ src, int n) {
  int idx = blockIdx.x * blockDim.x + threadIdx.x;
  if (idx < n) dst[idx] = (_Float16)src[idx];
}

__global__ void cat_cvt_kernel(_Float16* __restrict__ dst, const float* __restrict__ A,
                               const float* __restrict__ Bm, int K1, int K2, int total) {
  int idx = blockIdx.x * blockDim.x + threadIdx.x;
  if (idx >= total) return;
  int KK = K1 + K2;
  int n = idx / KK, k = idx % KK;
  dst[idx] = (_Float16)(k < K1 ? A[n * K1 + k] : Bm[n * K2 + (k - K1)]);
}

__global__ void padrows_cvt_kernel(_Float16* __restrict__ dst, const float* __restrict__ src,
                                   int Nvalid, int Kd, int total) {
  int idx = blockIdx.x * blockDim.x + threadIdx.x;
  if (idx >= total) return;
  int n = idx / Kd, k = idx % Kd;
  dst[idx] = (n < Nvalid) ? (_Float16)src[n * Kd + k] : (_Float16)0.0f;
}

__global__ void add_kernel(float* __restrict__ d, const float* __restrict__ a,
                           const float* __restrict__ b, int n) {
  int idx = blockIdx.x * blockDim.x + threadIdx.x;
  if (idx < n) d[idx] = a[idx] + b[idx];
}

__global__ void zerof_kernel(float* __restrict__ p, int n) {
  int idx = blockIdx.x * blockDim.x + threadIdx.x;
  if (idx < n) p[idx] = 0.0f;
}

// --------------------------------- attention --------------------------------------
__global__ void attn_kernel(const float* __restrict__ keys, const float* __restrict__ vals,
                            const float* __restrict__ h2, float* __restrict__ ctx) {
  __shared__ float qsh[E_];
  __shared__ float sw[S_];
  __shared__ float red[256];
  int b = blockIdx.x, tid = threadIdx.x;
  qsh[tid] = h2[b * E_ + tid];
  __syncthreads();

  const float* Kb = keys + (size_t)b * S_ * E_;
  float loc[4];
  float lmax = -1e30f;
#pragma unroll
  for (int i = 0; i < 4; ++i) {
    int s = tid + i * 256;
    const float* kr = Kb + (size_t)s * E_;
    float a0 = 0.f, a1 = 0.f, a2 = 0.f, a3 = 0.f;
    for (int e = 0; e < E_; e += 16) {
      float4 k0 = *reinterpret_cast<const float4*>(kr + e);
      float4 k1 = *reinterpret_cast<const float4*>(kr + e + 4);
      float4 k2 = *reinterpret_cast<const float4*>(kr + e + 8);
      float4 k3 = *reinterpret_cast<const float4*>(kr + e + 12);
      a0 += k0.x * qsh[e] + k0.y * qsh[e + 1] + k0.z * qsh[e + 2] + k0.w * qsh[e + 3];
      a1 += k1.x * qsh[e + 4] + k1.y * qsh[e + 5] + k1.z * qsh[e + 6] + k1.w * qsh[e + 7];
      a2 += k2.x * qsh[e + 8] + k2.y * qsh[e + 9] + k2.z * qsh[e + 10] + k2.w * qsh[e + 11];
      a3 += k3.x * qsh[e + 12] + k3.y * qsh[e + 13] + k3.z * qsh[e + 14] + k3.w * qsh[e + 15];
    }
    loc[i] = (a0 + a1 + a2 + a3) * 0.0625f;   // 1/sqrt(256)
    lmax = fmaxf(lmax, loc[i]);
  }
  red[tid] = lmax; __syncthreads();
  for (int st = 128; st > 0; st >>= 1) {
    if (tid < st) red[tid] = fmaxf(red[tid], red[tid + st]);
    __syncthreads();
  }
  float mx = red[0]; __syncthreads();

  float lsum = 0.0f;
#pragma unroll
  for (int i = 0; i < 4; ++i) {
    float ev = __expf(loc[i] - mx);
    sw[tid + i * 256] = ev;
    lsum += ev;
  }
  red[tid] = lsum; __syncthreads();
  for (int st = 128; st > 0; st >>= 1) {
    if (tid < st) red[tid] += red[tid + st];
    __syncthreads();
  }
  float inv = 1.0f / red[0];

  const float* Vb = vals + (size_t)b * S_ * E_;
  float p0 = 0.f, p1 = 0.f, p2 = 0.f, p3 = 0.f;
  for (int s = 0; s < S_; s += 4) {
    p0 += sw[s]     * Vb[(size_t)s * E_ + tid];
    p1 += sw[s + 1] * Vb[(size_t)(s + 1) * E_ + tid];
    p2 += sw[s + 2] * Vb[(size_t)(s + 2) * E_ + tid];
    p3 += sw[s + 3] * Vb[(size_t)(s + 3) * E_ + tid];
  }
  ctx[b * E_ + tid] = (p0 + p1 + p2 + p3) * inv;
}

// ------------------------------------ launch --------------------------------------
static inline size_t al256(size_t x) { return (x + 255) & ~(size_t)255; }

extern "C" void kernel_launch(void* const* d_in, const int* in_sizes, int n_in,
                              void* d_out, int out_size, void* d_ws, size_t ws_size,
                              hipStream_t stream) {
  (void)in_sizes; (void)n_in; (void)out_size; (void)ws_size;
  const float* seq_emb = (const float*)d_in[0];
  const int*   lens    = (const int*)d_in[1];
  const int*   batch_y = (const int*)d_in[2];
  const float* emb_W   = (const float*)d_in[3];
  const float* Wk      = (const float*)d_in[4];
  const float* bk      = (const float*)d_in[5];
  const float* Wv      = (const float*)d_in[6];
  const float* bv      = (const float*)d_in[7];
  const float* W_ih1   = (const float*)d_in[8];
  const float* W_hh1   = (const float*)d_in[9];
  const float* b_ih1   = (const float*)d_in[10];
  const float* b_hh1   = (const float*)d_in[11];
  const float* W_ih2   = (const float*)d_in[12];
  const float* W_hh2   = (const float*)d_in[13];
  const float* b_ih2   = (const float*)d_in[14];
  const float* b_hh2   = (const float*)d_in[15];
  const float* Wt      = (const float*)d_in[16];
  const float* bt      = (const float*)d_in[17];
  const float* cdn_b   = (const float*)d_in[18];
  float* out = (float*)d_out;

  // ---- workspace carve ----
  char* p = (char*)d_ws;
  auto take = [&](size_t bytes) { char* r = p; p += al256(bytes); return r; };
  _Float16* masked_h = (_Float16*)take((size_t)B_ * S_ * E_ * 2);
  float*    keys     = (float*)   take((size_t)B_ * S_ * E_ * 4);
  float*    values   = (float*)   take((size_t)B_ * S_ * E_ * 4);
  _Float16* W1cat_h  = (_Float16*)take((size_t)2048 * 1024 * 2);
  _Float16* W2cat_h  = (_Float16*)take((size_t)1024 * 768 * 2);
  _Float16* Wt_h     = (_Float16*)take((size_t)256 * 512 * 2);
  _Float16* Wk_h     = (_Float16*)take((size_t)256 * 256 * 2);
  _Float16* Wv_h     = (_Float16*)take((size_t)256 * 256 * 2);
  _Float16* embpad_h = (_Float16*)take((size_t)VPAD * 256 * 2);
  float*    b1sum    = (float*)   take(2048 * 4);
  float*    b2sum    = (float*)   take(1024 * 4);
  // states: c1 | h1a | h1b | h2a | h2b | c2 contiguous for one-shot zeroing
  float*    c1       = (float*)   take((size_t)B_ * 512 * 4);
  float*    h1a      = (float*)   take((size_t)B_ * 512 * 4);
  float*    h1b      = (float*)   take((size_t)B_ * 512 * 4);
  float*    h2a      = (float*)   take((size_t)B_ * 256 * 4);
  float*    h2b      = (float*)   take((size_t)B_ * 256 * 4);
  float*    c2       = (float*)   take((size_t)B_ * 256 * 4);
  float*    ctx      = (float*)   take((size_t)B_ * 256 * 4);
  _Float16* t_h      = (_Float16*)take((size_t)B_ * 256 * 2);

  // ---- one-time setup (re-run each call; deterministic) ----
  mask_cvt_kernel<<<(B_ * S_ * E_) / 256, 256, 0, stream>>>(seq_emb, lens, masked_h);
  cvt_kernel<<<256, 256, 0, stream>>>(Wk_h, Wk, 256 * 256);
  cvt_kernel<<<256, 256, 0, stream>>>(Wv_h, Wv, 256 * 256);
  cat_cvt_kernel<<<8192, 256, 0, stream>>>(W1cat_h, W_ih1, W_hh1, 512, 512, 2048 * 1024);
  cat_cvt_kernel<<<3072, 256, 0, stream>>>(W2cat_h, W_ih2, W_hh2, 512, 256, 1024 * 768);
  cvt_kernel<<<512, 256, 0, stream>>>(Wt_h, Wt, 256 * 512);
  padrows_cvt_kernel<<<1024, 256, 0, stream>>>(embpad_h, emb_W, V_, 256, VPAD * 256);
  add_kernel<<<8, 256, 0, stream>>>(b1sum, b_ih1, b_hh1, 2048);
  add_kernel<<<4, 256, 0, stream>>>(b2sum, b_ih2, b_hh2, 1024);
  {
    int nstate = B_ * 512 * 3 + B_ * 256 * 3;   // c1,h1a,h1b,h2a,h2b,c2
    zerof_kernel<<<(nstate + 255) / 256, 256, 0, stream>>>(c1, nstate);
  }

  // keys / values projections (one-time, WMMA): M=65536, N=256, K=256
  {
    dim3 g(4, 1024), blk(128);
    gemm_wmma_xt<<<g, blk, 0, stream>>>(masked_h, 256, Wk_h, 256, bk, keys, nullptr,
                                        256, 256, 256, 0);
    gemm_wmma_xt<<<g, blk, 0, stream>>>(masked_h, 256, Wv_h, 256, bv, values, nullptr,
                                        256, 256, 256, 0);
  }

  // initial context: attend with q = 0 (h2a just zeroed)
  attn_kernel<<<B_, 256, 0, stream>>>(keys, values, h2a, ctx);

  // ---- 599 sequential decode steps (5 kernels each) ----
  for (int step = 0; step < NSTEPS; ++step) {
    const float* h1o = (step & 1) ? h1b : h1a;
    float*       h1n = (step & 1) ? h1a : h1b;
    const float* h2o = (step & 1) ? h2b : h2a;
    float*       h2n = (step & 1) ? h2a : h2b;

    lstm1_fused_kernel<<<32, 128, 0, stream>>>(emb_W, batch_y, step, ctx, h1o,
                                               W1cat_h, b1sum, c1, h1n);
    lstm2_fused_kernel<<<16, 128, 0, stream>>>(h1n, h2o, W2cat_h, b2sum, c2, h2n);
    attn_kernel<<<B_, 256, 0, stream>>>(keys, values, h2n, ctx);
    cdn1_fused_kernel<<<4, 128, 0, stream>>>(h2n, ctx, Wt_h, bt, t_h);
    gemm_wmma_xt<<<dim3(16, 1), 128, 0, stream>>>(t_h, 256, embpad_h, 256, cdn_b,
                                                  out + (size_t)step * V_, nullptr,
                                                  (long)NSTEPS * V_, V_, 256, 0);
  }
}
